// XLSTMNetwork_67680094650781
// MI455X (gfx1250) — matrix operand order repaired and checked by reading, not compiled, not run on hardware
//
#include <hip/hip_runtime.h>
#include <hip/hip_bf16.h>

typedef _Float16 h16;
typedef __attribute__((ext_vector_type(16))) _Float16 v16h;
typedef __attribute__((ext_vector_type(8)))  float    v8f;

#define BB 64
#define TT 512
#define HH 256
#define BT (BB*TT)

// ---------------------------------------------------------------- WMMA helpers

__device__ inline v8f wmma32(v16h a, v16h b, v8f c) {
  // D = A(16x32 f16) * B(32x16 f16) + C(f32)
  return __builtin_amdgcn_wmma_f32_16x16x32_f16(false, a, false, b, (short)0, c, false, false);
}

// A fragment: rows [m0,m0+16), cols [k0,k0+32) of row-major f16 matrix (ld = lda).
// Layout per ISA 7.12.2 (16-bit A 16x32): lane holds M row; halves hold K with
// the documented interleave (VGPR0..3 = K 0..7 (+8 for hi lanes), VGPR4..7 = K16..23(+8)).
__device__ inline v16h load_a_frag(const h16* A, int lda, int m0, int k0, int lane) {
  const int row = m0 + (lane & 15);
  const int g8  = (lane >> 4) << 3;  // 0 or 8
  const h16* p = A + (size_t)row * lda + k0;
  v16h a;
#pragma unroll
  for (int v = 0; v < 8; ++v) {
    const int kp = (v < 4 ? 2 * v : 16 + 2 * (v - 4)) + g8;
    a[2 * v]     = p[kp];
    a[2 * v + 1] = p[kp + 1];
  }
  return a;
}

// B fragment from row-major [K,N] matrix: lane = N column, elements = K in table order.
__device__ inline v16h load_b_frag(const h16* Bm, int ldb, int k0, int n0, int lane) {
  const h16* p = Bm + (size_t)k0 * ldb + n0 + (lane & 15);
  const int g8 = (lane >> 4) << 3;
  v16h b;
#pragma unroll
  for (int i = 0; i < 16; ++i) b[i] = p[(size_t)((i < 8 ? i : 8 + i) + g8) * ldb];
  return b;
}

// B fragment where the logical B = M^T for row-major M[N,K] (used for Q@K^T).
__device__ inline v16h load_bt_frag(const h16* Bm, int ldb, int k0, int n0, int lane) {
  v16h b;
#pragma unroll
  for (int i = 0; i < 16; ++i) b[i] = Bm[(size_t)(n0 + (lane & 15)) * ldb + k0 + (i < 8 ? i : 8 + i) + ((lane >> 4) << 3)];
  return b;
}

__device__ inline float gelu_f(float x) {
  return 0.5f * x * (1.0f + erff(x * 0.70710678118654752f));
}

// ---------------------------------------------------------------- small kernels

__global__ void zero_u32_kernel(unsigned* p, int n) {
  int i = blockIdx.x * blockDim.x + threadIdx.x;
  if (i < n) p[i] = 0u;
}

__global__ void cast_f32_f16_kernel(const float* in, h16* out, size_t n) {
  size_t i = (size_t)blockIdx.x * blockDim.x + threadIdx.x;
  if (i < n) out[i] = (h16)in[i];
}

// W [N,K] f32 -> Wt [K,N] f16
__global__ void wcastT_kernel(const float* W, h16* Wt, int N, int K) {
  int idx = blockIdx.x * blockDim.x + threadIdx.x;
  if (idx >= N * K) return;
  int n = idx / K, k = idx % K;
  Wt[(size_t)k * N + n] = (h16)W[idx];
}

// ---------------------------------------------------------------- generic WMMA GEMM
// out[M,N] = A[M,K](f16) @ Bt[K,N](f16) + bias + residual ; store f32 and/or f16.
// Block = 256 threads (8 waves), tile 64x64, wave = 16x32.
__global__ void gemm_f16_kernel(const h16* __restrict__ A, const h16* __restrict__ Bt,
                                const float* __restrict__ bias,
                                const float* __restrict__ residual,
                                float* __restrict__ outF, h16* __restrict__ outH,
                                int K, int N) {
  const int lane = threadIdx.x & 31;
  const int wave = threadIdx.x >> 5;
  const int m0    = blockIdx.x * 64 + (wave & 3) * 16;
  const int nbase = blockIdx.y * 64 + (wave >> 2) * 32;
  v8f acc0 = {}, acc1 = {};
  for (int k0 = 0; k0 < K; k0 += 32) {
    v16h a  = load_a_frag(A, K, m0, k0, lane);
    v16h b0 = load_b_frag(Bt, N, k0, nbase, lane);
    v16h b1 = load_b_frag(Bt, N, k0, nbase + 16, lane);
    acc0 = wmma32(a, b0, acc0);
    acc1 = wmma32(a, b1, acc1);
  }
  const int col0 = nbase + (lane & 15);
  const int col1 = col0 + 16;
  const int rb   = m0 + 8 * (lane >> 4);
  const float bi0 = bias ? bias[col0] : 0.f;
  const float bi1 = bias ? bias[col1] : 0.f;
#pragma unroll
  for (int r = 0; r < 8; ++r) {
    const size_t row = (size_t)(rb + r);
    float v0 = acc0[r] + bi0, v1 = acc1[r] + bi1;
    if (residual) { v0 += residual[row * N + col0]; v1 += residual[row * N + col1]; }
    if (outF) { outF[row * N + col0] = v0; outF[row * N + col1] = v1; }
    if (outH) { outH[row * N + col0] = (h16)v0; outH[row * N + col1] = (h16)v1; }
  }
}

// ---------------------------------------------------------------- LayerNorm (wave per row)
__global__ void ln_kernel(const float* __restrict__ in, const float* __restrict__ g,
                          const float* __restrict__ be, float* __restrict__ outF,
                          h16* __restrict__ outH, long rows, int dim, int dogelu) {
  const int lane = threadIdx.x & 31;
  const int wave = threadIdx.x >> 5;
  const long row = (long)blockIdx.x * 8 + wave;
  if (row >= rows) return;
  const float* p = in + (size_t)row * dim;
  const int per = dim >> 5;   // dim/32, <= 8 here
  float vals[8];
  float s = 0.f, s2 = 0.f;
  for (int i = 0; i < per; ++i) {
    float v = p[lane + i * 32];
    vals[i] = v; s += v; s2 += v * v;
  }
  for (int off = 16; off; off >>= 1) {
    s  += __shfl_xor(s, off, 32);
    s2 += __shfl_xor(s2, off, 32);
  }
  const float mu  = s / dim;
  const float var = s2 / dim - mu * mu;
  const float inv = rsqrtf(var + 1e-5f);
  for (int i = 0; i < per; ++i) {
    const int c = lane + i * 32;
    float v = (vals[i] - mu) * inv * g[c] + be[c];
    if (dogelu) v = gelu_f(v);
    if (outF) outF[(size_t)row * dim + c] = v;
    if (outH) outH[(size_t)row * dim + c] = (h16)v;
  }
}

// ---------------------------------------------------------------- persistent sLSTM scan
__device__ inline void grid_barrier(unsigned* counter, unsigned nwg, unsigned& gen) {
  __syncthreads();
  __threadfence();
  ++gen;
  if (threadIdx.x == 0) {
    atomicAdd(counter, 1u);
    const unsigned target = gen * nwg;
    while (*((volatile unsigned*)counter) < target) __builtin_amdgcn_s_sleep(1);
  }
  __syncthreads();
  __threadfence();
}

// 4 WGs; WG k owns h columns [64k,64k+64). Each step: gates = xg_t + h@WhhT + bhh,
// state update, residual cur += h_t, ping-pong h in f16 for next step's WMMA GEMM.
__global__ void scan_kernel(const h16* __restrict__ xg,     // [B,T,4H] f16
                            const h16* __restrict__ WhhT,   // [H,4H]  f16
                            const float* __restrict__ bhh,  // [4H]
                            float* __restrict__ cur,        // [B,T,H] residual accumulate
                            h16* __restrict__ hbuf,         // [2][B*H] f16 ping-pong
                            float* __restrict__ st,         // h,c,n,m each [B*H] (d_out region)
                            unsigned* counter) {
  const int k    = blockIdx.x;          // 0..3
  const int lane = threadIdx.x & 31;
  const int wave = threadIdx.x >> 5;
  __shared__ float gl[64 * 256];        // gates: [b][g*64 + col], 64 KB LDS

  // zero this WG's state slice (lives in d_out) and its slice of h ping buffer 0
  for (int i = threadIdx.x; i < 64 * 64; i += 256) {
    const int b = i >> 6, c = i & 63;
    const int s = b * HH + k * 64 + c;
    st[s] = 0.f; st[BB * HH + s] = 0.f; st[2 * BB * HH + s] = 0.f; st[3 * BB * HH + s] = 0.f;
    hbuf[s] = (h16)0.f;
  }
  // warm the WGP cache with step 0's gate inputs (global_prefetch_b8, no LOADcnt)
  {
    const int pb = threadIdx.x >> 2;    // batch row 0..63
    const int pg = threadIdx.x & 3;     // gate 0..3
    __builtin_prefetch(xg + ((size_t)pb * TT) * (4 * HH) + pg * HH + k * 64, 0, 3);
  }
  unsigned gen = 0;
  grid_barrier(counter, 4, gen);

  const int g  = wave >> 1;             // gate 0..3 (z,i,f,o)
  const int nh = wave & 1;              // 32-col half within the 64-col slice

  for (int t = 0; t < TT; ++t) {
    const h16* hin  = hbuf + (size_t)(t & 1) * (BB * HH);
    h16*       hout = hbuf + (size_t)((t + 1) & 1) * (BB * HH);
    const int wcol = g * HH + k * 64 + nh * 32;   // column in [4H] gate space
    const float bb0 = bhh[wcol + (lane & 15)];
    const float bb1 = bhh[wcol + 16 + (lane & 15)];
#pragma unroll 1
    for (int m = 0; m < 4; ++m) {
      v8f c0 = {}, c1 = {};
      for (int kk = 0; kk < HH; kk += 32) {
        v16h a  = load_a_frag(hin, HH, m * 16, kk, lane);
        v16h b0 = load_b_frag(WhhT, 4 * HH, kk, wcol, lane);
        v16h b1 = load_b_frag(WhhT, 4 * HH, kk, wcol + 16, lane);
        c0 = wmma32(a, b0, c0);
        c1 = wmma32(a, b1, c1);
      }
      const int rb  = m * 16 + 8 * (lane >> 4);
      const int lc0 = nh * 32 + (lane & 15);
#pragma unroll
      for (int r = 0; r < 8; ++r) {
        const int b = rb + r;
        const size_t xrow = ((size_t)b * TT + t) * (4 * HH);
        gl[b * 256 + g * 64 + lc0]      = c0[r] + bb0 + (float)xg[xrow + wcol + (lane & 15)];
        gl[b * 256 + g * 64 + lc0 + 16] = c1[r] + bb1 + (float)xg[xrow + wcol + 16 + (lane & 15)];
      }
    }
    __syncthreads();
    // prefetch next step's xg lines for this WG's slice while we do the state update;
    // the scan is a serial dependency chain, so this hides the gate-input latency.
    if (t + 1 < TT) {
      const int pb = threadIdx.x >> 2;  // batch row 0..63
      const int pg = threadIdx.x & 3;   // gate 0..3
      __builtin_prefetch(xg + ((size_t)pb * TT + (t + 1)) * (4 * HH) + pg * HH + k * 64, 0, 3);
    }
    // elementwise sLSTM update for owned slice
    for (int i = threadIdx.x; i < 64 * 64; i += 256) {
      const int b = i >> 6, c = i & 63;
      const float z  = gl[b * 256 + c];
      const float ii = gl[b * 256 + 64 + c];
      const float ff = gl[b * 256 + 128 + c];
      const float oo = gl[b * 256 + 192 + c];
      const int s = b * HH + k * 64 + c;
      const float mo = st[3 * BB * HH + s];
      const float co = st[BB * HH + s];
      const float no = st[2 * BB * HH + s];
      const float mt = fmaxf(ff + mo, ii);
      const float ip = __expf(ii - mt);
      const float fp = __expf(ff + mo - mt);
      const float ct = fp * co + ip * tanhf(z);
      const float nt = fp * no + ip;
      const float ht = (1.f / (1.f + __expf(-oo))) * (ct / (nt + 1e-6f));
      st[s] = ht; st[BB * HH + s] = ct; st[2 * BB * HH + s] = nt; st[3 * BB * HH + s] = mt;
      hout[s] = (h16)ht;
      cur[((size_t)b * TT + t) * HH + k * 64 + c] += ht;  // residual out = x_seq + h
    }
    grid_barrier(counter, 4, gen);
  }
}

// ---------------------------------------------------------------- flash attention
// grid: B*NH*(T/64) WGs of 256 threads; each handles one (b,head,q-tile of 64).
__global__ void attn_kernel(const h16* __restrict__ qkv,   // [B*T, 768] f16 (q|k|v)
                            h16* __restrict__ ctx) {       // [B*T, 256] f16
  const int wg   = blockIdx.x;
  const int qt   = wg & 7;
  const int head = (wg >> 3) & 3;
  const int bb   = wg >> 5;
  const int lane = threadIdx.x & 31;
  const int wave = threadIdx.x >> 5;
  __shared__ h16 qs[4096], ks[4096], vs[4096], ps[4096];
  __shared__ float Ss[4096];
  __shared__ float mrow[64], lrow[64], srow[64];

  for (int i = threadIdx.x; i < 4096; i += 256) {
    const int r = i >> 6, c = i & 63;
    qs[i] = qkv[((size_t)(bb * TT + qt * 64 + r)) * 768 + head * 64 + c];
  }
  if (threadIdx.x < 64) { mrow[threadIdx.x] = -1e30f; lrow[threadIdx.x] = 0.f; }
  v8f o0 = {}, o1 = {};
  const int m0  = (wave & 3) * 16;
  const int n0w = (wave >> 2) * 32;
  __syncthreads();

  for (int kt = 0; kt < 8; ++kt) {
    for (int i = threadIdx.x; i < 4096; i += 256) {
      const int r = i >> 6, c = i & 63;
      const size_t base = ((size_t)(bb * TT + kt * 64 + r)) * 768 + head * 64 + c;
      ks[i] = qkv[base + 256];
      vs[i] = qkv[base + 512];
    }
    __syncthreads();
    // S = (Q @ K^T) / sqrt(dh)
    {
      v8f s0 = {}, s1 = {};
      for (int k0 = 0; k0 < 64; k0 += 32) {
        v16h a  = load_a_frag(qs, 64, m0, k0, lane);
        v16h b0 = load_bt_frag(ks, 64, k0, n0w, lane);
        v16h b1 = load_bt_frag(ks, 64, k0, n0w + 16, lane);
        s0 = wmma32(a, b0, s0);
        s1 = wmma32(a, b1, s1);
      }
      const int col0 = n0w + (lane & 15);
      const int rb   = m0 + 8 * (lane >> 4);
#pragma unroll
      for (int r = 0; r < 8; ++r) {
        Ss[(rb + r) * 64 + col0]      = s0[r] * 0.125f;
        Ss[(rb + r) * 64 + col0 + 16] = s1[r] * 0.125f;
      }
    }
    __syncthreads();
    // online softmax row update (threads 0..63, one row each)
    if (threadIdx.x < 64) {
      const int r = threadIdx.x;
      float rowmax = -1e30f;
      for (int c = 0; c < 64; ++c) rowmax = fmaxf(rowmax, Ss[r * 64 + c]);
      const float mnew = fmaxf(mrow[r], rowmax);
      const float corr = __expf(mrow[r] - mnew);
      float sum = 0.f;
      for (int c = 0; c < 64; ++c) {
        const float p = __expf(Ss[r * 64 + c] - mnew);
        ps[r * 64 + c] = (h16)p;
        sum += p;
      }
      lrow[r] = lrow[r] * corr + sum;
      mrow[r] = mnew;
      srow[r] = corr;
    }
    __syncthreads();
    // rescale O, accumulate P @ V
    {
      const int rb = m0 + 8 * (lane >> 4);
#pragma unroll
      for (int r = 0; r < 8; ++r) {
        const float sc = srow[rb + r];
        o0[r] *= sc; o1[r] *= sc;
      }
      for (int k0 = 0; k0 < 64; k0 += 32) {
        v16h a  = load_a_frag(ps, 64, m0, k0, lane);
        v16h b0 = load_b_frag(vs, 64, k0, n0w, lane);
        v16h b1 = load_b_frag(vs, 64, k0, n0w + 16, lane);
        o0 = wmma32(a, b0, o0);
        o1 = wmma32(a, b1, o1);
      }
    }
    __syncthreads();
  }
  const int col0 = n0w + (lane & 15);
  const int rb   = m0 + 8 * (lane >> 4);
#pragma unroll
  for (int r = 0; r < 8; ++r) {
    const int row = rb + r;
    const float inv = 1.f / lrow[row];
    const size_t o = ((size_t)(bb * TT + qt * 64 + row)) * HH + head * 64;
    ctx[o + col0]      = (h16)(o0[r] * inv);
    ctx[o + col0 + 16] = (h16)(o1[r] * inv);
  }
}

// ---------------------------------------------------------------- small fp32 linear
// out[rows,N] = [A1|A2] @ W[N,K1+K2]^T + bias ; optional exact GELU.
__global__ void slinear_kernel(const float* __restrict__ A1, int K1,
                               const float* __restrict__ A2, int K2,
                               const float* __restrict__ W, const float* __restrict__ bias,
                               float* __restrict__ out, int rows, int N, int dogelu) {
  const int idx = blockIdx.x * blockDim.x + threadIdx.x;
  if (idx >= rows * N) return;
  const int r = idx / N, n = idx % N;
  const float* w = W + (size_t)n * (K1 + K2);
  float acc = bias ? bias[n] : 0.f;
  for (int k = 0; k < K1; ++k) acc += A1[(size_t)r * K1 + k] * w[k];
  if (A2) for (int k = 0; k < K2; ++k) acc += A2[(size_t)r * K2 + k] * w[K1 + k];
  out[idx] = dogelu ? gelu_f(acc) : acc;
}

// attention pooling: softmax over T of scores, weighted sum of attn rows.
__global__ void pool_kernel(const float* __restrict__ attn, const float* __restrict__ sc,
                            float* __restrict__ ctxv) {
  __shared__ float red[256];
  __shared__ float sw[TT];
  const int b = blockIdx.x, tid = threadIdx.x;
  float mx = -1e30f;
  for (int t = tid; t < TT; t += 256) mx = fmaxf(mx, sc[b * TT + t]);
  red[tid] = mx; __syncthreads();
  for (int s = 128; s; s >>= 1) { if (tid < s) red[tid] = fmaxf(red[tid], red[tid + s]); __syncthreads(); }
  const float m = red[0];
  __syncthreads();
  float sum = 0.f;
  for (int t = tid; t < TT; t += 256) { const float e = __expf(sc[b * TT + t] - m); sw[t] = e; sum += e; }
  red[tid] = sum; __syncthreads();
  for (int s = 128; s; s >>= 1) { if (tid < s) red[tid] += red[tid + s]; __syncthreads(); }
  const float inv = 1.f / red[0];
  __syncthreads();
  const int h = tid;  // HH == 256
  float acc = 0.f;
  for (int t = 0; t < TT; ++t) acc += sw[t] * attn[((size_t)b * TT + t) * HH + h];
  ctxv[b * HH + h] = acc * inv;
}

// actor head: logits = a @ actor_w2^T + b ; softmax over 3.
__global__ void actor_head_kernel(const float* __restrict__ a, const float* __restrict__ W,
                                  const float* __restrict__ bias, float* __restrict__ out) {
  const int b = threadIdx.x;
  if (b >= 64) return;
  float l[3];
  for (int n = 0; n < 3; ++n) {
    float acc = bias[n];
    for (int k = 0; k < 64; ++k) acc += a[b * 64 + k] * W[n * 64 + k];
    l[n] = acc;
  }
  const float mx = fmaxf(l[0], fmaxf(l[1], l[2]));
  const float e0 = __expf(l[0] - mx), e1 = __expf(l[1] - mx), e2 = __expf(l[2] - mx);
  const float s = e0 + e1 + e2;
  out[b * 3 + 0] = e0 / s; out[b * 3 + 1] = e1 / s; out[b * 3 + 2] = e2 / s;
}

// ---------------------------------------------------------------- launch

extern "C" void kernel_launch(void* const* d_in, const int* in_sizes, int n_in,
                              void* d_out, int out_size, void* d_ws, size_t ws_size,
                              hipStream_t stream) {
  (void)in_sizes; (void)n_in; (void)out_size; (void)ws_size;
  // Input order: setup_inputs() dict insertion order, params nested in insertion order.
  const float* x        = (const float*)d_in[0];   // [B,T,64]
  const float* info     = (const float*)d_in[1];   // [B,13]
  const float* in_w     = (const float*)d_in[2];   // [256,64]
  const float* in_b     = (const float*)d_in[3];
  const float* ln_g[2]  = { (const float*)d_in[4],  (const float*)d_in[10] };
  const float* ln_b[2]  = { (const float*)d_in[5],  (const float*)d_in[11] };
  const float* Wih[2]   = { (const float*)d_in[6],  (const float*)d_in[12] };
  const float* bih[2]   = { (const float*)d_in[7],  (const float*)d_in[13] };
  const float* Whh[2]   = { (const float*)d_in[8],  (const float*)d_in[14] };
  const float* bhh[2]   = { (const float*)d_in[9],  (const float*)d_in[15] };
  const float* attn_in_w  = (const float*)d_in[16];
  const float* attn_in_b  = (const float*)d_in[17];
  const float* attn_out_w = (const float*)d_in[18];
  const float* attn_out_b = (const float*)d_in[19];
  const float* attn_ln_g  = (const float*)d_in[20];
  const float* attn_ln_b  = (const float*)d_in[21];
  const float* pool_w     = (const float*)d_in[22];
  const float* pool_b     = (const float*)d_in[23];
  const float* info_w1    = (const float*)d_in[24];
  const float* info_b1    = (const float*)d_in[25];
  const float* info_ln_g  = (const float*)d_in[26];
  const float* info_ln_b  = (const float*)d_in[27];
  const float* info_w2    = (const float*)d_in[28];
  const float* info_b2    = (const float*)d_in[29];
  const float* trunk_w1   = (const float*)d_in[30];
  const float* trunk_b1   = (const float*)d_in[31];
  const float* trunk_ln1_g= (const float*)d_in[32];
  const float* trunk_ln1_b= (const float*)d_in[33];
  const float* trunk_w2   = (const float*)d_in[34];
  const float* trunk_b2   = (const float*)d_in[35];
  const float* trunk_ln2_g= (const float*)d_in[36];
  const float* trunk_ln2_b= (const float*)d_in[37];
  const float* actor_w1   = (const float*)d_in[38];
  const float* actor_b1   = (const float*)d_in[39];
  const float* actor_w2   = (const float*)d_in[40];
  const float* actor_b2   = (const float*)d_in[41];
  const float* critic_w1  = (const float*)d_in[42];
  const float* critic_b1  = (const float*)d_in[43];
  const float* critic_ln_g= (const float*)d_in[44];
  const float* critic_ln_b= (const float*)d_in[45];
  const float* critic_w2  = (const float*)d_in[46];
  const float* critic_b2  = (const float*)d_in[47];

  float* outf = (float*)d_out;   // [192 probs][64 value][layer0 h,c,n,m][layer1 h,c,n,m]

  // ---- workspace layout
  char* ws = (char*)d_ws;
  size_t off = 0;
  auto take = [&](size_t bytes) { size_t r = off; off += (bytes + 255) & ~(size_t)255; return r; };
  const size_t H_IN   = 0;                  // f16 halves offsets inside w16 region
  const size_t H_WIH0 = 16384;
  const size_t H_WHH0 = 16384 + 262144;
  const size_t H_WIH1 = 16384 + 2 * 262144;
  const size_t H_WHH1 = 16384 + 3 * 262144;
  const size_t H_AIN  = 16384 + 4 * 262144;
  const size_t H_AOUT = H_AIN + 196608;
  const size_t W16_TOTAL = H_AOUT + 65536;  // 1,327,104 halves
  size_t o_w16  = take(W16_TOTAL * 2);
  size_t o_x16  = take((size_t)BT * 64 * 2);
  size_t o_cur  = take((size_t)BT * HH * 4);
  size_t o_a16  = take((size_t)BT * HH * 2);
  size_t o_big  = take((size_t)BT * 4 * HH * 2);  // xg16, later qkv16 (768) + ctx16 (256)
  size_t o_attn = take((size_t)BT * HH * 4);
  size_t o_hbuf = take((size_t)2 * BB * HH * 2);
  size_t o_pool = take((size_t)BT * 4);
  size_t o_ctxv = take((size_t)BB * HH * 4);
  size_t o_ie1  = take(64 * 64 * 4);
  size_t o_ie2  = take(64 * 64 * 4);
  size_t o_t1p  = take(64 * 256 * 4);
  size_t o_t1   = take(64 * 256 * 4);
  size_t o_t2p  = take(64 * 128 * 4);
  size_t o_t2   = take(64 * 128 * 4);
  size_t o_ab   = take(64 * 64 * 4);
  size_t o_cb1  = take(64 * 64 * 4);
  size_t o_cb2  = take(64 * 64 * 4);
  size_t o_sync = take(256);

  h16* w16   = (h16*)(ws + o_w16);
  h16* x16   = (h16*)(ws + o_x16);
  float* cur = (float*)(ws + o_cur);
  h16* a16   = (h16*)(ws + o_a16);
  h16* big16 = (h16*)(ws + o_big);
  float* attnb = (float*)(ws + o_attn);
  h16* hbuf  = (h16*)(ws + o_hbuf);
  float* pools = (float*)(ws + o_pool);
  float* ctxv  = (float*)(ws + o_ctxv);
  unsigned* sync = (unsigned*)(ws + o_sync);
  h16* ctx16 = big16 + (size_t)BT * 768;   // 768 + 256 = 1024 halves/row total

  const dim3 blk(256);

  // 0) zero the barrier counters (fresh every call -> deterministic / graph-safe)
  hipLaunchKernelGGL(zero_u32_kernel, dim3(1), dim3(64), 0, stream, sync, 64);

  // 1) cast x to f16
  hipLaunchKernelGGL(cast_f32_f16_kernel, dim3((BT * 64 + 255) / 256), blk, 0, stream,
                     x, x16, (size_t)BT * 64);

  // 2) transpose+cast weights to f16 [K,N]
  hipLaunchKernelGGL(wcastT_kernel, dim3((256 * 64 + 255) / 256), blk, 0, stream, in_w, w16 + H_IN, 256, 64);
  hipLaunchKernelGGL(wcastT_kernel, dim3((1024 * 256 + 255) / 256), blk, 0, stream, Wih[0], w16 + H_WIH0, 1024, 256);
  hipLaunchKernelGGL(wcastT_kernel, dim3((1024 * 256 + 255) / 256), blk, 0, stream, Whh[0], w16 + H_WHH0, 1024, 256);
  hipLaunchKernelGGL(wcastT_kernel, dim3((1024 * 256 + 255) / 256), blk, 0, stream, Wih[1], w16 + H_WIH1, 1024, 256);
  hipLaunchKernelGGL(wcastT_kernel, dim3((1024 * 256 + 255) / 256), blk, 0, stream, Whh[1], w16 + H_WHH1, 1024, 256);
  hipLaunchKernelGGL(wcastT_kernel, dim3((768 * 256 + 255) / 256), blk, 0, stream, attn_in_w, w16 + H_AIN, 768, 256);
  hipLaunchKernelGGL(wcastT_kernel, dim3((256 * 256 + 255) / 256), blk, 0, stream, attn_out_w, w16 + H_AOUT, 256, 256);

  // 3) input projection: cur = x @ in_w^T + in_b     [BT,64]x[64,256]
  hipLaunchKernelGGL(gemm_f16_kernel, dim3(BT / 64, HH / 64), blk, 0, stream,
                     x16, w16 + H_IN, in_b, (const float*)nullptr, cur, (h16*)nullptr, 64, HH);

  // 4) two sLSTM layers
  const size_t hW[2] = { H_WIH0, H_WIH1 };
  const size_t hR[2] = { H_WHH0, H_WHH1 };
  for (int l = 0; l < 2; ++l) {
    hipLaunchKernelGGL(ln_kernel, dim3(BT / 8), blk, 0, stream,
                       cur, ln_g[l], ln_b[l], (float*)nullptr, a16, (long)BT, HH, 0);
    hipLaunchKernelGGL(gemm_f16_kernel, dim3(BT / 64, (4 * HH) / 64), blk, 0, stream,
                       a16, w16 + hW[l], bih[l], (const float*)nullptr,
                       (float*)nullptr, big16, HH, 4 * HH);
    hipLaunchKernelGGL(scan_kernel, dim3(4), blk, 0, stream,
                       big16, w16 + hR[l], bhh[l], cur, hbuf,
                       outf + 256 + (size_t)l * 4 * BB * HH, sync + l);
  }

  // 5) attention: qkv = cur @ attn_in_w^T + b (f16)
  hipLaunchKernelGGL(cast_f32_f16_kernel, dim3((BT * HH + 255) / 256), blk, 0, stream,
                     cur, a16, (size_t)BT * HH);
  hipLaunchKernelGGL(gemm_f16_kernel, dim3(BT / 64, 768 / 64), blk, 0, stream,
                     a16, w16 + H_AIN, attn_in_b, (const float*)nullptr,
                     (float*)nullptr, big16, HH, 768);
  hipLaunchKernelGGL(attn_kernel, dim3(BB * 4 * (TT / 64)), blk, 0, stream, big16, ctx16);
  // out-proj + residual, then LN into cur (cur becomes attn_out)
  hipLaunchKernelGGL(gemm_f16_kernel, dim3(BT / 64, HH / 64), blk, 0, stream,
                     ctx16, w16 + H_AOUT, attn_out_b, cur, attnb, (h16*)nullptr, HH, HH);
  hipLaunchKernelGGL(ln_kernel, dim3(BT / 8), blk, 0, stream,
                     attnb, attn_ln_g, attn_ln_b, cur, (h16*)nullptr, (long)BT, HH, 0);

  // 6) attention pooling -> context [B,H]
  hipLaunchKernelGGL(slinear_kernel, dim3((BT + 255) / 256), blk, 0, stream,
                     cur, HH, (const float*)nullptr, 0, pool_w, pool_b, pools, BT, 1, 0);
  hipLaunchKernelGGL(pool_kernel, dim3(BB), blk, 0, stream, cur, pools, ctxv);

  // 7) info branch: ie = gelu(linear2(gelu(ln(linear1(info)))))
  float* ie1 = (float*)(ws + o_ie1);
  float* ie2 = (float*)(ws + o_ie2);
  hipLaunchKernelGGL(slinear_kernel, dim3((64 * 64 + 255) / 256), blk, 0, stream,
                     info, 13, (const float*)nullptr, 0, info_w1, info_b1, ie1, 64, 64, 0);
  hipLaunchKernelGGL(ln_kernel, dim3(8), blk, 0, stream,
                     ie1, info_ln_g, info_ln_b, ie2, (h16*)nullptr, 64L, 64, 1);
  hipLaunchKernelGGL(slinear_kernel, dim3((64 * 64 + 255) / 256), blk, 0, stream,
                     ie2, 64, (const float*)nullptr, 0, info_w2, info_b2, ie1, 64, 64, 1);

  // 8) trunk
  float* t1p = (float*)(ws + o_t1p); float* t1 = (float*)(ws + o_t1);
  float* t2p = (float*)(ws + o_t2p); float* t2 = (float*)(ws + o_t2);
  hipLaunchKernelGGL(slinear_kernel, dim3((64 * 256 + 255) / 256), blk, 0, stream,
                     ctxv, 256, ie1, 64, trunk_w1, trunk_b1, t1p, 64, 256, 0);
  hipLaunchKernelGGL(ln_kernel, dim3(8), blk, 0, stream,
                     t1p, trunk_ln1_g, trunk_ln1_b, t1, (h16*)nullptr, 64L, 256, 1);
  hipLaunchKernelGGL(slinear_kernel, dim3((64 * 128 + 255) / 256), blk, 0, stream,
                     t1, 256, (const float*)nullptr, 0, trunk_w2, trunk_b2, t2p, 64, 128, 0);
  hipLaunchKernelGGL(ln_kernel, dim3(8), blk, 0, stream,
                     t2p, trunk_ln2_g, trunk_ln2_b, t2, (h16*)nullptr, 64L, 128, 1);

  // 9) actor head -> d_out[0:192]
  float* ab = (float*)(ws + o_ab);
  hipLaunchKernelGGL(slinear_kernel, dim3((64 * 64 + 255) / 256), blk, 0, stream,
                     t2, 128, (const float*)nullptr, 0, actor_w1, actor_b1, ab, 64, 64, 1);
  hipLaunchKernelGGL(actor_head_kernel, dim3(1), dim3(64), 0, stream, ab, actor_w2, actor_b2, outf);

  // 10) critic head -> d_out[192:256]
  float* cb1 = (float*)(ws + o_cb1);
  float* cb2 = (float*)(ws + o_cb2);
  hipLaunchKernelGGL(slinear_kernel, dim3((64 * 64 + 255) / 256), blk, 0, stream,
                     t2, 128, (const float*)nullptr, 0, critic_w1, critic_b1, cb1, 64, 64, 0);
  hipLaunchKernelGGL(ln_kernel, dim3(8), blk, 0, stream,
                     cb1, critic_ln_g, critic_ln_b, cb2, (h16*)nullptr, 64L, 64, 1);
  hipLaunchKernelGGL(slinear_kernel, dim3(1), dim3(64), 0, stream,
                     cb2, 64, (const float*)nullptr, 0, critic_w2, critic_b2, outf + 192, 64, 1, 0);
}
